// MultiHeadSelfAttention_77421080478079
// MI455X (gfx1250) — matrix operand order repaired and checked by reading, not compiled, not run on hardware
//
#include <hip/hip_runtime.h>
#include <math.h>

typedef __bf16 bf16;
typedef __attribute__((ext_vector_type(16))) bf16  v16bf;
typedef __attribute__((ext_vector_type(8)))  bf16  bf16x8;
typedef __attribute__((ext_vector_type(4)))  bf16  bf16x4;
typedef __attribute__((ext_vector_type(8)))  float v8f;

#define B_   4
#define N_   2048
#define D_   1024
#define H_   16
#define DH_  64
#define BN_  (B_ * N_)
#define KB_  64            // flash-attention key block

// ---------------------------------------------------------------------------
// WMMA helper: D = A(16x32 bf16) * B(32x16 bf16) + C(16x16 f32)
// ---------------------------------------------------------------------------
static __device__ __forceinline__ v8f wmma_bf16(v16bf a, v16bf b, v8f c) {
  return __builtin_amdgcn_wmma_f32_16x16x32_bf16(
      /*neg_a=*/false, a, /*neg_b=*/false, b,
      /*c_mod=*/(short)0, c, /*reuse_a=*/false, /*reuse_b=*/false);
}

// A-fragment (16x32, 16-bit): lane L holds row M=L%16; with half=L/16 the 16
// elements are K = half*16 + e, e=0..15 -> two b128 loads.
static __device__ __forceinline__ v16bf load_frag_a(const bf16* rowk, int hf) {
  bf16x8 lo = *(const bf16x8*)(rowk + hf * 16);
  bf16x8 hi = *(const bf16x8*)(rowk + hf * 16 + 8);
  return __builtin_shufflevector(lo, hi, 0,1,2,3,4,5,6,7,8,9,10,11,12,13,14,15);
}

// B-fragment (32x16, 16-bit): lane L holds col N=L%16; elements are
// K = half*16 + e, e=0..15 -> 32 contiguous bytes (caller adds half*16).
static __device__ __forceinline__ v16bf load_frag_b(const bf16* colk) {
  bf16x8 lo = *(const bf16x8*)(colk);
  bf16x8 hi = *(const bf16x8*)(colk + 8);
  return __builtin_shufflevector(lo, hi, 0,1,2,3,4,5,6,7,8,9,10,11,12,13,14,15);
}

static __device__ __forceinline__ float rowmax16(float v) {
  v = fmaxf(v, __shfl_xor(v, 1, 32));
  v = fmaxf(v, __shfl_xor(v, 2, 32));
  v = fmaxf(v, __shfl_xor(v, 4, 32));
  v = fmaxf(v, __shfl_xor(v, 8, 32));
  return v;
}

// ---------------------------------------------------------------------------
// fp32 -> bf16 conversion (vectorized x4)
// ---------------------------------------------------------------------------
__global__ __launch_bounds__(256) void cvt_f32_bf16(const float* __restrict__ s,
                                                    bf16* __restrict__ d, int n) {
  int i = (blockIdx.x * blockDim.x + threadIdx.x) * 4;
  if (i < n) {
    float4 v = *(const float4*)(s + i);
    bf16x4 o;
    o[0] = (bf16)v.x; o[1] = (bf16)v.y; o[2] = (bf16)v.z; o[3] = (bf16)v.w;
    *(bf16x4*)(d + i) = o;
  }
}

// ---------------------------------------------------------------------------
// GEMM: Y = (A[M][K] @ W[N][K]^T + bias) * outscale. One wave: 32x64 tile.
// MODE 0: bf16 row-major out (+bias, *outscale)   (Q pre-scaled, K)
// MODE 1: bf16 head-transposed Vt out (+bias): Vt[b][h][d][n]
// MODE 2: f32 row-major out, no bias              (final projection)
// ---------------------------------------------------------------------------
template <int MODE>
__global__ __launch_bounds__(256) void gemm_xwT(const bf16* __restrict__ A,
                                                const bf16* __restrict__ W,
                                                const float* __restrict__ bias,
                                                float outscale,
                                                bf16* __restrict__ outb,
                                                float* __restrict__ outf) {
  const int K = D_;
  const int NT = D_ / 64;
  int wave = (int)((blockIdx.x * blockDim.x + threadIdx.x) >> 5);
  int lane = (int)(threadIdx.x & 31);
  int hf = lane >> 4;        // wave half (0: lanes 0-15, 1: lanes 16-31)
  int c  = lane & 15;        // column / row-within-16
  int tm = wave / NT, tn = wave % NT;
  int row0 = tm * 32, col0 = tn * 64;

  v8f acc[2][4];
#pragma unroll
  for (int i = 0; i < 2; ++i)
#pragma unroll
    for (int j = 0; j < 4; ++j)
#pragma unroll
      for (int r = 0; r < 8; ++r) acc[i][j][r] = 0.0f;

  const bf16* a0p = A + (size_t)(row0 + c) * K;
  const bf16* a1p = A + (size_t)(row0 + 16 + c) * K;
  const bf16* bbp = W + (size_t)(col0 + c) * K + hf * 16;

  for (int k0 = 0; k0 < K; k0 += 32) {
    v16bf a0 = load_frag_a(a0p + k0, hf);
    v16bf a1 = load_frag_a(a1p + k0, hf);
#pragma unroll
    for (int ni = 0; ni < 4; ++ni) {
      v16bf b = load_frag_b(bbp + (size_t)ni * 16 * K + k0);
      acc[0][ni] = wmma_bf16(a0, b, acc[0][ni]);
      acc[1][ni] = wmma_bf16(a1, b, acc[1][ni]);
    }
  }

#pragma unroll
  for (int mi = 0; mi < 2; ++mi) {
#pragma unroll
    for (int ni = 0; ni < 4; ++ni) {
      int col = col0 + ni * 16 + c;
      float bv = (MODE == 2) ? 0.0f : bias[col];
      int rbase = row0 + mi * 16 + 8 * hf;   // C layout: lane holds M = r + 8*half
      if (MODE == 0) {
#pragma unroll
        for (int r = 0; r < 8; ++r)
          outb[(size_t)(rbase + r) * D_ + col] =
              (bf16)((acc[mi][ni][r] + bv) * outscale);
      } else if (MODE == 1) {
        int hh = col >> 6, d = col & 63;
        int bidx = rbase >> 11, n0 = rbase & (N_ - 1);   // 8 contiguous tokens
        bf16x8 pk;
#pragma unroll
        for (int r = 0; r < 8; ++r) pk[r] = (bf16)(acc[mi][ni][r] + bv);
        *(bf16x8*)(outb + (size_t)((bidx * H_ + hh) * DH_ + d) * N_ + n0) = pk;
      } else {
#pragma unroll
        for (int r = 0; r < 8; ++r)
          outf[(size_t)(rbase + r) * D_ + col] = acc[mi][ni][r];
      }
    }
  }
}

// ---------------------------------------------------------------------------
// Flash attention, base-2 online softmax. One wave owns 16 query rows of one
// (b,h); key blocks of 64. Q was pre-scaled by Dh^-0.5 * log2(e), so
// weights = 2^(s - m). Row sums accumulate via WMMA against an all-ones
// B fragment (lacc behaves exactly like the O accumulators under rescaling).
// ---------------------------------------------------------------------------
__global__ __launch_bounds__(256) void flash_attn(const bf16* __restrict__ Q,
                                                  const bf16* __restrict__ Kr,
                                                  const bf16* __restrict__ Vt,
                                                  bf16* __restrict__ O) {
  __shared__ __align__(16) bf16 Plds[8][16][KB_];   // 2KB per wave
  int wib  = (int)(threadIdx.x >> 5);
  int gw   = (int)blockIdx.x * 8 + wib;
  int lane = (int)(threadIdx.x & 31);
  int hf = lane >> 4, c = lane & 15;

  int qt = gw & 127;               // N_/16 query tiles
  int hh = (gw >> 7) & (H_ - 1);
  int b  = gw >> 11;
  int q0 = qt * 16;

  const bf16* qrow = Q + (size_t)(b * N_ + q0 + c) * D_ + hh * DH_;
  v16bf aq0 = load_frag_a(qrow + 0, hf);
  v16bf aq1 = load_frag_a(qrow + 32, hf);

  v16bf ones;
#pragma unroll
  for (int e = 0; e < 16; ++e) ones[e] = (bf16)1.0f;

  v8f o[4], lacc;
  float m[8], al[8];
#pragma unroll
  for (int dt = 0; dt < 4; ++dt)
#pragma unroll
    for (int r = 0; r < 8; ++r) o[dt][r] = 0.0f;
#pragma unroll
  for (int r = 0; r < 8; ++r) { m[r] = -1e30f; lacc[r] = 0.0f; }

  const bf16* kbase = Kr + (size_t)(b * N_) * D_ + hh * DH_;
  const bf16* vhead = Vt + (size_t)((b * H_ + hh) * DH_) * N_;
  bf16* myP = &Plds[wib][0][0];

  for (int j = 0; j < N_ / KB_; ++j) {
    int kb0 = j * KB_;
    v8f s[4];
#pragma unroll
    for (int st = 0; st < 4; ++st) {
      const bf16* krow = kbase + (size_t)(kb0 + st * 16 + c) * D_ + hf * 16;
      v16bf bk0 = load_frag_b(krow);        // d = 0..31 (lane half picks 16)
      v16bf bk1 = load_frag_b(krow + 32);   // d = 32..63
      v8f z;
#pragma unroll
      for (int r = 0; r < 8; ++r) z[r] = 0.0f;
      s[st] = wmma_bf16(aq0, bk0, wmma_bf16(aq1, bk1, z));
    }

#pragma unroll
    for (int r = 0; r < 8; ++r) {
      float v0 = s[0][r], v1 = s[1][r], v2 = s[2][r], v3 = s[3][r];
      float mx = rowmax16(fmaxf(fmaxf(v0, v1), fmaxf(v2, v3)));  // 64 keys
      float mnew = fmaxf(m[r], mx);
      al[r] = exp2f(m[r] - mnew);
      m[r]  = mnew;
      float p0 = exp2f(v0 - mnew);
      float p1 = exp2f(v1 - mnew);
      float p2 = exp2f(v2 - mnew);
      float p3 = exp2f(v3 - mnew);
      int row = r + 8 * hf;                 // row-major 16x64 P tile in LDS
      myP[row * KB_ + c]      = (bf16)p0;
      myP[row * KB_ + 16 + c] = (bf16)p1;
      myP[row * KB_ + 32 + c] = (bf16)p2;
      myP[row * KB_ + 48 + c] = (bf16)p3;
    }

#pragma unroll
    for (int r = 0; r < 8; ++r) lacc[r] *= al[r];
#pragma unroll
    for (int dt = 0; dt < 4; ++dt)
#pragma unroll
      for (int r = 0; r < 8; ++r) o[dt][r] *= al[r];

    // Re-read P as A-fragments (same-wave LDS ops are in-order).
    v16bf pa0 = load_frag_a(myP + c * KB_, hf);        // keys kb0+0..31
    v16bf pa1 = load_frag_a(myP + c * KB_ + 32, hf);   // keys kb0+32..63

    // Row sums: P @ ones -> every column of lacc holds sum_k P[row][k].
    lacc = wmma_bf16(pa0, ones, lacc);
    lacc = wmma_bf16(pa1, ones, lacc);

#pragma unroll
    for (int dt = 0; dt < 4; ++dt) {
      const bf16* vrow = vhead + (size_t)(dt * 16 + c) * N_ + kb0 + hf * 16;
      v16bf vb0 = load_frag_b(vrow);
      v16bf vb1 = load_frag_b(vrow + 32);
      o[dt] = wmma_bf16(pa0, vb0, o[dt]);
      o[dt] = wmma_bf16(pa1, vb1, o[dt]);
    }
  }

#pragma unroll
  for (int r = 0; r < 8; ++r) {
    float inv = 1.0f / lacc[r];
    size_t rowoff = (size_t)(b * N_ + q0 + 8 * hf + r) * D_ + hh * DH_;
#pragma unroll
    for (int dt = 0; dt < 4; ++dt)
      O[rowoff + dt * 16 + c] = (bf16)(o[dt][r] * inv);
  }
}

// ---------------------------------------------------------------------------
extern "C" void kernel_launch(void* const* d_in, const int* in_sizes, int n_in,
                              void* d_out, int out_size, void* d_ws, size_t ws_size,
                              hipStream_t stream) {
  const float* x  = (const float*)d_in[0];
  const float* Wq = (const float*)d_in[1];
  const float* bq = (const float*)d_in[2];
  const float* Wk = (const float*)d_in[3];
  const float* bk = (const float*)d_in[4];
  const float* Wv = (const float*)d_in[5];
  const float* bv = (const float*)d_in[6];
  const float* Wo = (const float*)d_in[7];
  float* out = (float*)d_out;

  char* ws = (char*)d_ws;
  const size_t SZ_ACT = (size_t)BN_ * D_ * sizeof(bf16);  // 16 MB
  const size_t SZ_W   = (size_t)D_ * D_ * sizeof(bf16);   // 2 MB
  bf16* xb  = (bf16*)ws;  ws += SZ_ACT;
  bf16* qb  = (bf16*)ws;  ws += SZ_ACT;
  bf16* kb  = (bf16*)ws;  ws += SZ_ACT;
  bf16* vtb = (bf16*)ws;  ws += SZ_ACT;   // Vt[b][h][d][n]
  bf16* ab  = (bf16*)ws;  ws += SZ_ACT;   // attention output (bf16)
  bf16* wqb = (bf16*)ws;  ws += SZ_W;
  bf16* wkb = (bf16*)ws;  ws += SZ_W;
  bf16* wvb = (bf16*)ws;  ws += SZ_W;
  bf16* wob = (bf16*)ws;  ws += SZ_W;

  // Dh^-0.5 * log2(e): folds both the softmax scale and the base-2 exp
  // conversion into the Q projection.
  const float QSCALE = 0.125f * 1.4426950408889634f;

  // 1) convert activations + weights to bf16
  cvt_f32_bf16<<<(BN_ * D_) / 1024, 256, 0, stream>>>(x, xb, BN_ * D_);
  cvt_f32_bf16<<<(D_ * D_) / 1024, 256, 0, stream>>>(Wq, wqb, D_ * D_);
  cvt_f32_bf16<<<(D_ * D_) / 1024, 256, 0, stream>>>(Wk, wkb, D_ * D_);
  cvt_f32_bf16<<<(D_ * D_) / 1024, 256, 0, stream>>>(Wv, wvb, D_ * D_);
  cvt_f32_bf16<<<(D_ * D_) / 1024, 256, 0, stream>>>(Wo, wob, D_ * D_);

  // 2) QKV projections: 4096 waves of 32x64 tiles -> 512 blocks x 8 waves
  const int GEMM_BLOCKS = (BN_ / 32) * (D_ / 64) / 8;
  gemm_xwT<0><<<GEMM_BLOCKS, 256, 0, stream>>>(xb, wqb, bq, QSCALE, qb, nullptr);
  gemm_xwT<0><<<GEMM_BLOCKS, 256, 0, stream>>>(xb, wkb, bk, 1.0f,   kb, nullptr);
  gemm_xwT<1><<<GEMM_BLOCKS, 256, 0, stream>>>(xb, wvb, bv, 1.0f,   vtb, nullptr);

  // 3) flash attention: B*H*(N/16) = 8192 waves -> 1024 blocks x 8 waves
  flash_attn<<<(B_ * H_ * (N_ / 16)) / 8, 256, 0, stream>>>(qb, kb, vtb, ab);

  // 4) output projection (fp32 out, no bias)
  gemm_xwT<2><<<GEMM_BLOCKS, 256, 0, stream>>>(ab, wob, nullptr, 1.0f, nullptr, out);
}
